// CFDErrorInterpolateOld_28767690948644
// MI455X (gfx1250) — matrix operand order repaired and checked by reading, not compiled, not run on hardware
//
#include <hip/hip_runtime.h>
#include <math.h>

typedef __attribute__((ext_vector_type(16))) _Float16 v16h;
typedef __attribute__((ext_vector_type(8)))  float    v8f;
typedef __attribute__((ext_vector_type(4)))  unsigned u32x4;
typedef __attribute__((ext_vector_type(8)))  int      i32x8;
typedef __attribute__((ext_vector_type(4)))  int      i32x4;

#if defined(__has_builtin)
#  if __has_builtin(__builtin_amdgcn_tensor_load_to_lds) && __has_builtin(__builtin_amdgcn_s_wait_tensorcnt)
#    define HAVE_TDM 1
#  endif
#endif
#ifndef HAVE_TDM
#  define HAVE_TDM 0
#endif

__device__ __forceinline__ void atomic_max_f32(float* addr, float v) {
  if (v >= 0.0f) atomicMax((int*)addr, __float_as_int(v));
  else           atomicMin((unsigned int*)addr, __float_as_uint(v));
}

#if HAVE_TDM
// Issue one TDM 2-D tile DMA: rows x cols f32, contiguous, global -> LDS.
// D# per CDNA5 ISA: group0 = {count=1 | lds_addr | global_addr | type=2},
// group1 = {data_size=4B, tensor_dim0/1, tile_dim0/1, dim0 stride}.
__device__ __forceinline__ void tdm_load_2d_f32(float* lds_dst, const float* g, int rows, int cols) {
  unsigned lds_off = (unsigned)(unsigned long long)lds_dst;      // LDS aperture: addr[31:0]
  unsigned long long ga = (unsigned long long)g;
  u32x4 g0;
  g0[0] = 1u;                                                    // count=1, user mode, no gather
  g0[1] = lds_off;                                               // lds_addr [63:32]
  g0[2] = (unsigned)(ga & 0xFFFFFFFFull);                        // global_addr[31:0]
  g0[3] = (unsigned)((ga >> 32) & 0x01FFFFFFull) | 0x80000000u;  // global_addr[56:32], type=2
  i32x8 g1;
  g1[0] = 0x20000;                                 // workgroup_mask=0, data_size=2 (4 bytes)
  g1[1] = (cols & 0xFFFF) << 16;                   // tensor_dim0 low16  @ [63:48]
  g1[2] = ((rows & 0xFFFF) << 16);                 // tensor_dim0 hi16=0, tensor_dim1 low16
  g1[3] = (cols & 0xFFFF) << 16;                   // tensor_dim1 hi16=0, tile_dim0
  g1[4] = rows & 0xFFFF;                           // tile_dim1; tile_dim2=0 (2-D)
  g1[5] = cols;                                    // tensor_dim0_stride low32 (elements)
  g1[6] = 0;                                       // stride hi16, dim1_stride low16
  g1[7] = 0;
  i32x4 z4 = {0, 0, 0, 0};
#if __clang_major__ >= 23
  i32x8 z8 = {0, 0, 0, 0, 0, 0, 0, 0};
  __builtin_amdgcn_tensor_load_to_lds(g0, g1, z4, z4, z8, 0);
#else
  __builtin_amdgcn_tensor_load_to_lds(g0, g1, z4, z4, 0);
#endif
}
#endif

// Stage ROWS x COLS f32 weight matrix into LDS (TDM if available), then barrier.
template <int ROWS, int COLS>
__device__ __forceinline__ void stage_weights(float* ws, const float* g) {
#if HAVE_TDM
  if ((threadIdx.x >> 5) == 0) {                 // one wave issues the DMA (EXEC-independent op)
    tdm_load_2d_f32(ws, g, ROWS, COLS);
    __builtin_amdgcn_s_wait_tensorcnt(0);
  }
#else
  for (int i = threadIdx.x; i < ROWS * COLS; i += blockDim.x) ws[i] = g[i];
#endif
  __syncthreads();
}

// ---------------- elementwise / utility kernels ----------------
__global__ void k_fill_u32(unsigned* __restrict__ p, unsigned v, long n) {
  long stride = (long)gridDim.x * blockDim.x;
  for (long i = (long)blockIdx.x * blockDim.x + threadIdx.x; i < n; i += stride) p[i] = v;
}

__global__ void k_fixmax(float* __restrict__ p, long n) {
  long stride = (long)gridDim.x * blockDim.x;
  for (long i = (long)blockIdx.x * blockDim.x + threadIdx.x; i < n; i += stride) {
    float v = p[i];
    p[i] = (v == v && fabsf(v) <= 3.0e38f) ? v : 0.0f;  // !finite -> 0
  }
}

__global__ void k_add(const float* __restrict__ a, const float* __restrict__ b,
                      float* __restrict__ y, long n) {
  long stride = (long)gridDim.x * blockDim.x;
  for (long i = (long)blockIdx.x * blockDim.x + threadIdx.x; i < n; i += stride)
    y[i] = a[i] + b[i];
}

__global__ void k_build_xc(const float* __restrict__ x1, const float* __restrict__ x2,
                           const float* __restrict__ t0, const float* __restrict__ t1,
                           float* __restrict__ xc, int n) {
  long total = (long)n * 64;
  long stride = (long)gridDim.x * blockDim.x;
  for (long i = (long)blockIdx.x * blockDim.x + threadIdx.x; i < total; i += stride) {
    long node = i >> 6; int k = (int)(i & 63);
    xc[node * 192 + k] = x1[i];
    float a = x2[i];               xc[node * 192 + 64 + k]  = a * a;
    float b0 = t0[i], b1 = t1[i];  xc[node * 192 + 128 + k] = b0 * b0 + b1 * b1;
  }
}

__global__ void k_build_xcat(const float* __restrict__ xl, const float* __restrict__ e,
                             float* __restrict__ xcat, int n) {
  long total = (long)n * 3;
  long stride = (long)gridDim.x * blockDim.x;
  for (long i = (long)blockIdx.x * blockDim.x + threadIdx.x; i < total; i += stride) {
    long node = i / 3; int c = (int)(i % 3);
    xcat[node * 6 + c]     = xl[i];
    xcat[node * 6 + 3 + c] = e[i];
  }
}

__global__ void k_deg_edges(const int* __restrict__ dst, float* __restrict__ deg, int nE) {
  int e = blockIdx.x * blockDim.x + threadIdx.x;
  if (e < nE) atomicAdd(&deg[dst[e]], 1.0f);
}

__global__ void k_rsqrt_inplace(float* __restrict__ p, int n) {
  int i = blockIdx.x * blockDim.x + threadIdx.x;
  if (i < n) p[i] = rsqrtf(p[i]);
}

// ---------------- WMMA EdgeConv: per-edge MLP + segment-max scatter ----------------
// 4 waves/block, 16 edges/wave. Weights TDM-staged to LDS once per block.
// A tile (f16) assembled in LDS from gathered [x_i, x_j - x_i]; D scattered with atomic max.
template <int CIN, int COUT>
__launch_bounds__(128)
__global__ void k_edgeconv_wmma(const float* __restrict__ x,
                                const int* __restrict__ src, const int* __restrict__ dst,
                                const float* __restrict__ Wm, const float* __restrict__ bias,
                                float* __restrict__ agg, int nE) {
  constexpr int KROWS = 2 * CIN;
  constexpr int KTOT = ((KROWS + 31) / 32) * 32;
  constexpr int KT = KTOT / 32;
  constexpr int NT = (COUT + 15) / 16;
  __shared__ _Float16 A[4 * 16 * KTOT];
  __shared__ float WS[KROWS * COUT];

  stage_weights<KROWS, COUT>(WS, Wm);  // TDM -> s_wait_tensorcnt -> barrier

  const int tid = threadIdx.x;
  const int wave = tid >> 5, lane = tid & 31;
  const int m = lane & 15, half = lane >> 4;
  const int base = blockIdx.x * 64 + wave * 16;
  const int e = base + m;
  const bool ev = e < nE;
  const int si = ev ? src[e] : 0;
  const int di = ev ? dst[e] : 0;
  const float* xi = x + (long)di * CIN;
  const float* xj = x + (long)si * CIN;
  _Float16* row = &A[(wave * 16 + m) * KTOT];
  if (half == 0) {
    for (int k = 0; k < CIN; k++) row[k] = ev ? (_Float16)xi[k] : (_Float16)0.0f;
  } else {
    for (int k = 0; k < CIN; k++) row[CIN + k] = ev ? (_Float16)(xj[k] - xi[k]) : (_Float16)0.0f;
    for (int k = KROWS; k < KTOT; k++) row[k] = (_Float16)0.0f;
  }
  __syncthreads();

  v8f acc[NT] = {};
#pragma unroll
  for (int t = 0; t < KT; ++t) {
    v16h a;
#pragma unroll
    for (int r = 0; r < 8; r++) {
      const int k0 = t * 32 + (r >> 2) * 16 + half * 8 + (r & 3) * 2;  // ISA A-frag layout
      a[2 * r]     = A[(wave * 16 + m) * KTOT + k0];
      a[2 * r + 1] = A[(wave * 16 + m) * KTOT + k0 + 1];
    }
    const int kb = t * 32 + half * 8;  // ISA B-frag: lane <-> N column, K in table order
#pragma unroll
    for (int nt = 0; nt < NT; ++nt) {
      v16h b;
#pragma unroll
      for (int r = 0; r < 8; r++) {
        const int k0 = kb + (r >> 2) * 16 + (r & 3) * 2;
        const int n0 = nt * 16 + m;
        float f0 = (k0     < KROWS && n0 < COUT) ? WS[k0 * COUT + n0]       : 0.0f;
        float f1 = (k0 + 1 < KROWS && n0 < COUT) ? WS[(k0 + 1) * COUT + n0] : 0.0f;
        b[2 * r]     = (_Float16)f0;
        b[2 * r + 1] = (_Float16)f1;
      }
      acc[nt] = __builtin_amdgcn_wmma_f32_16x16x32_f16(false, a, false, b, (short)0,
                                                       acc[nt], false, false);
    }
  }

  const int nl = lane & 15;
#pragma unroll
  for (int nt = 0; nt < NT; ++nt) {
    const int n = nt * 16 + nl;
    if (n < COUT) {
#pragma unroll
      for (int g = 0; g < 8; ++g) {
        const int mm = g + half * 8;  // ISA D layout
        const int ee = base + mm;
        if (ee < nE) {
          const int dd = dst[ee];
          atomic_max_f32(&agg[(long)dd * COUT + n], acc[nt][g] + bias[n]);
        }
      }
    }
  }
}

// ---------------- WMMA dense GEMM: Y = X @ W (+bias) ----------------
// 4 waves/block, 16 rows/wave; W TDM-staged to LDS once per block.
template <int CIN, int COUT>
__launch_bounds__(128)
__global__ void k_gemm_wmma(const float* __restrict__ X, const float* __restrict__ Wm,
                            const float* __restrict__ bias, float* __restrict__ Y, int M) {
  constexpr int KTOT = ((CIN + 31) / 32) * 32;
  constexpr int KT = KTOT / 32;
  constexpr int NT = (COUT + 15) / 16;
  __shared__ float WS[CIN * COUT];

  stage_weights<CIN, COUT>(WS, Wm);

  const int tid = threadIdx.x;
  const int wave = tid >> 5, lane = tid & 31;
  const int m = lane & 15, half = lane >> 4;
  const long row0 = (long)blockIdx.x * 64 + wave * 16;
  const long row = row0 + m;
  const bool rv = row < M;

  v8f acc[NT] = {};
#pragma unroll
  for (int t = 0; t < KT; ++t) {
    v16h a;
#pragma unroll
    for (int r = 0; r < 8; r++) {
      const int k0 = t * 32 + (r >> 2) * 16 + half * 8 + (r & 3) * 2;
      float f0 = (rv && k0     < CIN) ? X[row * CIN + k0]     : 0.0f;
      float f1 = (rv && k0 + 1 < CIN) ? X[row * CIN + k0 + 1] : 0.0f;
      a[2 * r]     = (_Float16)f0;
      a[2 * r + 1] = (_Float16)f1;
    }
    const int kb = t * 32 + half * 8;
#pragma unroll
    for (int nt = 0; nt < NT; ++nt) {
      v16h b;
#pragma unroll
      for (int r = 0; r < 8; r++) {
        const int k0 = kb + (r >> 2) * 16 + (r & 3) * 2;
        const int n0 = nt * 16 + m;
        float f0 = (k0     < CIN && n0 < COUT) ? WS[k0 * COUT + n0]       : 0.0f;
        float f1 = (k0 + 1 < CIN && n0 < COUT) ? WS[(k0 + 1) * COUT + n0] : 0.0f;
        b[2 * r]     = (_Float16)f0;
        b[2 * r + 1] = (_Float16)f1;
      }
      acc[nt] = __builtin_amdgcn_wmma_f32_16x16x32_f16(false, a, false, b, (short)0,
                                                       acc[nt], false, false);
    }
  }

  const int nl = lane & 15;
#pragma unroll
  for (int nt = 0; nt < NT; ++nt) {
    const int n = nt * 16 + nl;
    if (n < COUT) {
#pragma unroll
      for (int g = 0; g < 8; ++g) {
        const long rr = row0 + g + half * 8;
        if (rr < M) {
          float bv = bias ? bias[n] : 0.0f;
          Y[rr * COUT + n] = acc[nt][g] + bv;
        }
      }
    }
  }
}

// ---------------- GCN message passing ----------------
template <int C>
__global__ void k_gcn_self(const float* __restrict__ h, const float* __restrict__ dinv,
                           const float* __restrict__ b, float* __restrict__ out, int n) {
  long total = (long)n * C;
  long stride = (long)gridDim.x * blockDim.x;
  for (long i = (long)blockIdx.x * blockDim.x + threadIdx.x; i < total; i += stride) {
    long node = i / C; int c = (int)(i % C);
    float dv = dinv[node];
    out[i] = h[i] * dv * dv + b[c];
  }
}

template <int C>
__global__ void k_gcn_scatter(const float* __restrict__ h, const int* __restrict__ src,
                              const int* __restrict__ dst, const float* __restrict__ dinv,
                              float* __restrict__ out, int nE) {
  int e = blockIdx.x * blockDim.x + threadIdx.x;
  if (e >= nE) return;
  int s = src[e], d = dst[e];
  float cf = dinv[s] * dinv[d];
  const float* hs = h + (long)s * C;
  float* od = out + (long)d * C;
#pragma unroll
  for (int c = 0; c < C; c++) atomicAdd(&od[c], hs[c] * cf);
}

// ---------------- BatchNorm (training-style batch stats) ----------------
__global__ void k_bn_stats(const float* __restrict__ x, float* __restrict__ stats, int M, int C) {
  __shared__ float sh1[256], sh2[256];
  const int c = blockIdx.y;
  float s = 0.0f, s2 = 0.0f;
  for (int r = blockIdx.x * blockDim.x + threadIdx.x; r < M; r += gridDim.x * blockDim.x) {
    float v = x[(long)r * C + c];
    s += v; s2 += v * v;
  }
  sh1[threadIdx.x] = s; sh2[threadIdx.x] = s2;
  __syncthreads();
  for (int o = 128; o > 0; o >>= 1) {
    if (threadIdx.x < o) { sh1[threadIdx.x] += sh1[threadIdx.x + o]; sh2[threadIdx.x] += sh2[threadIdx.x + o]; }
    __syncthreads();
  }
  if (threadIdx.x == 0) {
    atomicAdd(&stats[c],     sh1[0]);
    atomicAdd(&stats[C + c], sh2[0]);
  }
}

__global__ void k_bn_relu_add(const float* __restrict__ x, const float* __restrict__ stats,
                              const float* __restrict__ gamma, const float* __restrict__ beta,
                              const float* __restrict__ identity, float* __restrict__ y,
                              int M, int C) {
  long total = (long)M * C;
  long stride = (long)gridDim.x * blockDim.x;
  float invM = 1.0f / (float)M;
  for (long i = (long)blockIdx.x * blockDim.x + threadIdx.x; i < total; i += stride) {
    int c = (int)(i % C);
    float mu = stats[c] * invM;
    float var = stats[C + c] * invM - mu * mu;
    float v = (x[i] - mu) * rsqrtf(var + 1e-5f) * gamma[c] + beta[c];
    v = fmaxf(v, 0.0f);
    y[i] = v + (identity ? identity[i] : 0.0f);
  }
}

// ---------------- brute-force kNN (K=32) + inverse-d2 interpolation ----------------
__launch_bounds__(128)
__global__ void k_knn_interp(const float* __restrict__ posL, const float* __restrict__ posH,
                             const float* __restrict__ xcat, float* __restrict__ xh,
                             int nL, int nH) {
  constexpr int KNN = 32, TILE = 1024, BT = 128;
  __shared__ float2 tile[TILE];
  __shared__ float dls[KNN * BT];
  __shared__ int   ils[KNN * BT];
  const int tid = threadIdx.x;
  const int node = blockIdx.x * BT + tid;
  float2 ph; ph.x = 0.0f; ph.y = 0.0f;
  if (node < nH) { ph.x = posH[2 * node]; ph.y = posH[2 * node + 1]; }
  for (int j = 0; j < KNN; j++) { dls[j * BT + tid] = 1e30f; ils[j * BT + tid] = 0; }
  float worst = 1e30f; int wslot = 0;

  for (int base = 0; base < nL; base += TILE) {
    __syncthreads();
    for (int t = tid; t < TILE; t += BT) {
      int p = base + t;
      float2 v;
      if (p < nL) { v.x = posL[2 * p]; v.y = posL[2 * p + 1]; }
      else        { v.x = 1e15f;       v.y = 1e15f; }
      tile[t] = v;
    }
    __syncthreads();
    if (base + TILE < nL) __builtin_prefetch(&posL[2 * (base + TILE) + 2 * tid], 0, 0);
    int lim = nL - base; if (lim > TILE) lim = TILE;
    for (int t = 0; t < lim; t++) {
      float dx = ph.x - tile[t].x, dy = ph.y - tile[t].y;
      float d2 = dx * dx + dy * dy;
      if (d2 < worst) {
        dls[wslot * BT + tid] = d2;
        ils[wslot * BT + tid] = base + t;
        float w = -1.0f; int ws = 0;
#pragma unroll
        for (int j = 0; j < KNN; j++) {
          float v = dls[j * BT + tid];
          if (v > w) { w = v; ws = j; }
        }
        worst = w; wslot = ws;
      }
    }
  }

  float wsum = 0.0f;
  float acc[6] = {0, 0, 0, 0, 0, 0};
  for (int j = 0; j < KNN; j++) {
    float d2 = dls[j * BT + tid];
    int id = ils[j * BT + tid];
    float w = 1.0f / fmaxf(d2, 1e-16f);
    wsum += w;
    const float* xv = xcat + (long)id * 6;
#pragma unroll
    for (int c = 0; c < 6; c++) acc[c] += w * xv[c];
  }
  if (node < nH) {
    float inv = 1.0f / wsum;
    for (int c = 0; c < 6; c++) xh[(long)node * 6 + c] = acc[c] * inv;
  }
}

// ================= host orchestration =================
extern "C" void kernel_launch(void* const* d_in, const int* in_sizes, int n_in,
                              void* d_out, int out_size, void* d_ws, size_t ws_size,
                              hipStream_t stream) {
  const float* x_l  = (const float*)d_in[0];
  const float* posL = (const float*)d_in[1];
  const float* posH = (const float*)d_in[2];
  const int*   eiL  = (const int*)d_in[3];
  const int*   eiH  = (const int*)d_in[4];
  const int NL = in_sizes[1] / 2, NH = in_sizes[2] / 2;
  const int EL = in_sizes[3] / 2, EH = in_sizes[4] / 2;
  const int *srcL = eiL, *dstL = eiL + EL;
  const int *srcH = eiH, *dstH = eiH + EH;

  // params: jax pytree flatten = sorted dict keys, tuples in order
  int pi = 5;
  auto F = [&](void) { return (const float*)d_in[pi++]; };
  const float *dec1_W = F(), *dec1_b = F(), *dec1_beta = F(), *dec1_gamma = F();
  const float *dec2_W = F(), *dec2_b = F(), *dec2_beta = F(), *dec2_gamma = F(), *dec2_rW = F(), *dec2_rb = F();
  const float *enc1_W = F(), *enc1_b = F(), *enc1_beta = F(), *enc1_gamma = F(), *enc1_rW = F(), *enc1_rb = F();
  const float *enc2_W = F(), *enc2_b = F(), *enc2_beta = F(), *enc2_gamma = F();
  const float *errc_W = F(), *errc_b = F();
  const float *err1_W = F(), *err1_b = F();
  const float *errf0_W = F(), *errf0_b = F();
  const float *errf1_W = F(), *errf1_b = F();
  const float *erri_W = F(), *erri_b = F();
  const float *err2_W = F(), *err2_b = F();
  const float *mp_W = F(), *mp_b = F();
  (void)n_in; (void)out_size; (void)ws_size;

  float* w = (float*)d_ws;
  size_t off = 0;
  auto alloc = [&](size_t ne) { float* p = w + off; off += ne; return p; };
  float* xcat = alloc((size_t)NL * 6);
  float* a0   = alloc((size_t)NL * 64);
  float* a1   = alloc((size_t)NL * 64);
  float* a2   = alloc((size_t)NL * 64);
  float* t0   = alloc((size_t)NL * 64);
  float* t1   = alloc((size_t)NL * 64);
  float* xc   = alloc((size_t)NL * 192);
  float* e3   = alloc((size_t)NL * 3);
  float* xh   = alloc((size_t)NH * 6);
  float* dinv = alloc((size_t)NH);
  float* hb   = alloc((size_t)NH * 64);
  float* gout = alloc((size_t)NH * 64);
  float* x1b  = alloc((size_t)NH * 64);
  float* x2b  = alloc((size_t)NH * 64);
  float* db   = alloc((size_t)NH * 64);
  float* d1b  = alloc((size_t)NH * 64);
  float* idb  = alloc((size_t)NH * 64);
  float* stats = alloc(128);

  auto gsz = [](long n, int b) {
    long g = (n + b - 1) / b;
    if (g > 1048576) g = 1048576;
    if (g < 1) g = 1;
    return (unsigned)g;
  };
  auto fillinf = [&](float* p, long n) { k_fill_u32<<<gsz(n, 256), 256, 0, stream>>>((unsigned*)p, 0xFF800000u, n); };
  auto fix     = [&](float* p, long n) { k_fixmax<<<gsz(n, 256), 256, 0, stream>>>(p, n); };
  const unsigned ecg = (unsigned)((EL + 63) / 64);   // 4 waves x 16 edges per block

  // ---- CFDError branch on low graph ----
  fillinf(a0, (long)NL * 64);
  k_edgeconv_wmma<2, 64><<<ecg, 128, 0, stream>>>(posL, srcL, dstL, erri_W, erri_b, a0, EL);
  fix(a0, (long)NL * 64);

  fillinf(a1, (long)NL * 64);
  k_edgeconv_wmma<64, 64><<<ecg, 128, 0, stream>>>(a0, srcL, dstL, err1_W, err1_b, a1, EL);
  fix(a1, (long)NL * 64);

  fillinf(a2, (long)NL * 64);
  k_edgeconv_wmma<64, 64><<<ecg, 128, 0, stream>>>(a0, srcL, dstL, err2_W, err2_b, a2, EL);
  fix(a2, (long)NL * 64);

  fillinf(t0, (long)NL * 64);
  k_edgeconv_wmma<64, 64><<<ecg, 128, 0, stream>>>(a0, srcL, dstL, errf0_W, errf0_b, t0, EL);
  fix(t0, (long)NL * 64);

  fillinf(t1, (long)NL * 64);
  k_edgeconv_wmma<64, 64><<<ecg, 128, 0, stream>>>(a0, srcL, dstL, errf1_W, errf1_b, t1, EL);
  fix(t1, (long)NL * 64);

  k_build_xc<<<gsz((long)NL * 64, 256), 256, 0, stream>>>(a1, a2, t0, t1, xc, NL);

  fillinf(e3, (long)NL * 3);
  k_edgeconv_wmma<192, 3><<<ecg, 128, 0, stream>>>(xc, srcL, dstL, errc_W, errc_b, e3, EL);
  fix(e3, (long)NL * 3);

  k_build_xcat<<<gsz((long)NL * 3, 256), 256, 0, stream>>>(x_l, e3, xcat, NL);

  // ---- kNN interpolate to fine mesh ----
  k_knn_interp<<<(NH + 127) / 128, 128, 0, stream>>>(posL, posH, xcat, xh, NL, NH);

  // ---- GCN normalization coefficients (in-degree + self loop) ----
  k_fill_u32<<<gsz(NH, 256), 256, 0, stream>>>((unsigned*)dinv, 0x3F800000u, NH);  // 1.0f
  k_deg_edges<<<gsz(EH, 256), 256, 0, stream>>>(dstH, dinv, EH);
  k_rsqrt_inplace<<<gsz(NH, 256), 256, 0, stream>>>(dinv, NH);

  const unsigned gg = (unsigned)((NH + 63) / 64);    // 4 waves x 16 rows per block
  auto bn_zero = [&](int C) { k_fill_u32<<<1, 256, 0, stream>>>((unsigned*)stats, 0u, 2 * C); };

  // enc1: xh(6) -> x1b, residual projection
  k_gemm_wmma<6, 64><<<gg, 128, 0, stream>>>(xh, enc1_rW, enc1_rb, idb, NH);
  k_gemm_wmma<6, 64><<<gg, 128, 0, stream>>>(xh, enc1_W, (const float*)nullptr, hb, NH);
  k_gcn_self<64><<<gsz((long)NH * 64, 256), 256, 0, stream>>>(hb, dinv, enc1_b, gout, NH);
  k_gcn_scatter<64><<<gsz(EH, 256), 256, 0, stream>>>(hb, srcH, dstH, dinv, gout, EH);
  bn_zero(64);
  k_bn_stats<<<dim3(64, 64), 256, 0, stream>>>(gout, stats, NH, 64);
  k_bn_relu_add<<<gsz((long)NH * 64, 256), 256, 0, stream>>>(gout, stats, enc1_gamma, enc1_beta, idb, x1b, NH, 64);

  // enc2: x1b -> x2b, identity = x1b
  k_gemm_wmma<64, 64><<<gg, 128, 0, stream>>>(x1b, enc2_W, (const float*)nullptr, hb, NH);
  k_gcn_self<64><<<gsz((long)NH * 64, 256), 256, 0, stream>>>(hb, dinv, enc2_b, gout, NH);
  k_gcn_scatter<64><<<gsz(EH, 256), 256, 0, stream>>>(hb, srcH, dstH, dinv, gout, EH);
  bn_zero(64);
  k_bn_stats<<<dim3(64, 64), 256, 0, stream>>>(gout, stats, NH, 64);
  k_bn_relu_add<<<gsz((long)NH * 64, 256), 256, 0, stream>>>(gout, stats, enc2_gamma, enc2_beta, x1b, x2b, NH, 64);

  // mp: gcn(x2b) then d = x1 + xm
  k_gemm_wmma<64, 64><<<gg, 128, 0, stream>>>(x2b, mp_W, (const float*)nullptr, hb, NH);
  k_gcn_self<64><<<gsz((long)NH * 64, 256), 256, 0, stream>>>(hb, dinv, mp_b, gout, NH);
  k_gcn_scatter<64><<<gsz(EH, 256), 256, 0, stream>>>(hb, srcH, dstH, dinv, gout, EH);
  k_add<<<gsz((long)NH * 64, 256), 256, 0, stream>>>(x1b, gout, db, (long)NH * 64);

  // dec1: db -> d1b, identity = db
  k_gemm_wmma<64, 64><<<gg, 128, 0, stream>>>(db, dec1_W, (const float*)nullptr, hb, NH);
  k_gcn_self<64><<<gsz((long)NH * 64, 256), 256, 0, stream>>>(hb, dinv, dec1_b, gout, NH);
  k_gcn_scatter<64><<<gsz(EH, 256), 256, 0, stream>>>(hb, srcH, dstH, dinv, gout, EH);
  bn_zero(64);
  k_bn_stats<<<dim3(64, 64), 256, 0, stream>>>(gout, stats, NH, 64);
  k_bn_relu_add<<<gsz((long)NH * 64, 256), 256, 0, stream>>>(gout, stats, dec1_gamma, dec1_beta, db, d1b, NH, 64);

  // dec2: d1b -> out (64 -> 3, residual projection), result straight into d_out
  k_gemm_wmma<64, 3><<<gg, 128, 0, stream>>>(d1b, dec2_rW, dec2_rb, idb, NH);
  k_gemm_wmma<64, 3><<<gg, 128, 0, stream>>>(d1b, dec2_W, (const float*)nullptr, hb, NH);
  k_gcn_self<3><<<gsz((long)NH * 3, 256), 256, 0, stream>>>(hb, dinv, dec2_b, gout, NH);
  k_gcn_scatter<3><<<gsz(EH, 256), 256, 0, stream>>>(hb, srcH, dstH, dinv, gout, EH);
  bn_zero(3);
  k_bn_stats<<<dim3(64, 3), 256, 0, stream>>>(gout, stats, NH, 3);
  k_bn_relu_add<<<gsz((long)NH * 3, 256), 256, 0, stream>>>(gout, stats, dec2_gamma, dec2_beta, idb, (float*)d_out, NH, 3);
}
